// MultiHeadSelectiveAttention_16183436772081
// MI455X (gfx1250) — matrix operand order repaired and checked by reading, not compiled, not run on hardware
//
#include <hip/hip_runtime.h>
#include <hip/hip_bf16.h>

// Dims
#define BB 8
#define QQ 64
#define SS 64
#define TT 64
#define HH 8
#define DHD 64
#define NHD 512
#define LAST 16          // only last-16 tokens matter (exact: exp(-1e6-max) == 0 in fp32)
#define NEGV -1.0e6f

#define LDAP 72          // padded LDS row length (elements): 36 dwords -> conflict-free banks
#define BUFE (64 * LDAP) // elements per staged 64-row matrix (4608)

typedef __attribute__((ext_vector_type(16))) __bf16       v16bf;
typedef __attribute__((ext_vector_type(8)))  float        v8f;
typedef __attribute__((ext_vector_type(8)))  unsigned int v8u;

// ---------------- helpers ----------------

__device__ __forceinline__ unsigned short f2bf(float f) {
  unsigned int x = __float_as_uint(f);
  if ((x & 0x7fffffffu) > 0x7f800000u) return (unsigned short)((x >> 16) | 0x40u); // NaN
  x += 0x7fffu + ((x >> 16) & 1u);   // round-to-nearest-even
  return (unsigned short)(x >> 16);
}

// A fragment: 16x32 bf16, row-major source (lda elements, even). contiguous k order (not the ISA table order):
// lane L: m = L&15 ; lanes<16 hold K=0..15 ; lanes>=16 hold K=16..31.
__device__ __forceinline__ v16bf load_a16x32(const unsigned short* A, int lda, int lane) {
  int m  = lane & 15;
  int d0 = (lane >> 4) << 3;                    // dword base: 0 or 8 (K 0 / 16)
  const unsigned int* p = (const unsigned int*)(A + m * lda);
  v8u r;
  r[0] = p[d0 + 0]; r[1] = p[d0 + 1]; r[2] = p[d0 + 2]; r[3] = p[d0 + 3];   // K base..base+7
  r[4] = p[d0 + 4]; r[5] = p[d0 + 5]; r[6] = p[d0 + 6]; r[7] = p[d0 + 7];   // K base+8..+15
  return __builtin_bit_cast(v16bf, r);
}

// B fragment: 32x16 bf16 (KxN), source given TRANSPOSED (Bt[n][k], ldb elements).
__device__ __forceinline__ v16bf load_bt32x16(const unsigned short* Bt, int ldb, int lane) {
  int n  = lane & 15;
  int d0 = (lane >> 4) << 3;                    // dword base: 0 or 8
  const unsigned int* p = (const unsigned int*)(Bt + n * ldb);
  v8u r;
#pragma unroll
  for (int i = 0; i < 8; ++i) r[i] = p[d0 + i];
  return __builtin_bit_cast(v16bf, r);
}

// B fragment from ROW-major B[k][n] (ldb elements): strided 16-bit gathers.
__device__ __forceinline__ v16bf load_b32x16_rm(const unsigned short* Bm, int ldb, int lane) {
  int n  = lane & 15;
  int kb = (lane >> 4) << 4;                    // 0 or 16
  v8u r;
#pragma unroll
  for (int i = 0; i < 8; ++i) {
    unsigned lo = Bm[(kb + 2 * i) * ldb + n];
    unsigned hi = Bm[(kb + 2 * i + 1) * ldb + n];
    r[i] = lo | (hi << 16);
  }
  return __builtin_bit_cast(v16bf, r);
}

__device__ __forceinline__ v8f wmma_bf16(v16bf a, v16bf b, v8f c) {
  return __builtin_amdgcn_wmma_f32_16x16x32_bf16(false, a, false, b, (short)0, c, false, false);
}

__device__ __forceinline__ v8f vzero8() {
  v8f c;
#pragma unroll
  for (int i = 0; i < 8; ++i) c[i] = 0.0f;
  return c;
}

// Async 16B global->LDS copy (CDNA5, tracked by ASYNCcnt). LDS byte offset is the
// low 32 bits of the generic shared pointer.
__device__ __forceinline__ void async_b128(const unsigned short* lds, const unsigned short* g) {
  unsigned loff = (unsigned)(unsigned long long)(uintptr_t)lds;
  unsigned long long ga = (unsigned long long)(uintptr_t)g;
  asm volatile("global_load_async_to_lds_b128 %0, %1, off" :: "v"(loff), "v"(ga) : "memory");
}
__device__ __forceinline__ void wait_async0() {
  asm volatile("s_wait_asynccnt 0x0" ::: "memory");
}

// ---------------- conversion kernels ----------------

// W[in=k][out=n] fp32 -> Wt[n][k] bf16 (512x512)
__global__ void k_weight_t(const float* __restrict__ W, unsigned short* __restrict__ Wt) {
  int i = blockIdx.x * blockDim.x + threadIdx.x;   // 0..262143
  int n = i >> 9, k = i & 511;
  Wt[n * 512 + k] = f2bf(W[k * 512 + n]);
}

__global__ void k_cvt(const float* __restrict__ in, unsigned short* __restrict__ out, int n) {
  int i = blockIdx.x * blockDim.x + threadIdx.x;
  if (i < n) out[i] = f2bf(in[i]);
}

// in [B*S, 64, 512] -> out [B*S, 16, 512] (last 16 token rows), bf16
__global__ void k_last16(const float* __restrict__ in, unsigned short* __restrict__ out) {
  int i = blockIdx.x * blockDim.x + threadIdx.x;   // out elem, total 8192*512
  int c = i & 511;
  int row = i >> 9;                 // bs*16 + tt
  int tt = row & 15, bs = row >> 4;
  out[i] = f2bf(in[(bs * 64 + (TT - LAST) + tt) * 512 + c]);
}

// ---------------- 64x64 macro-tile GEMM core: async-to-LDS, double-buffered ----------------
// 8 waves: wave w -> rows (w&3)*16, cols ((w>>2)*32) + {0,16}. K in chunks of 64.
// LDS rows padded to LDAP=72 elements (36 dwords) -> conflict-free bank access.
// BT=true : B panel is Bt[n][k] (64 rows x K, stride ldb)
// BT=false: B panel is B[k][n] row-major (K rows x 64, stride ldb)
template <bool BT>
__device__ __forceinline__ void gemm64(const unsigned short* Ablk, int lda,
                                       const unsigned short* Bblk, int ldb,
                                       int K, float scale,
                                       unsigned short* Cbf, float* Cf, int ldc,
                                       unsigned short* smem) {
  int tid = threadIdx.x, wave = tid >> 5, lane = tid & 31;
  int rw = wave & 3, cw = (wave >> 2) << 5;
  v8f acc0 = vzero8(), acc1 = vzero8();
  int nch = K >> 6;

  auto stage = [&](int ch) {
    unsigned short* As = smem + (ch & 1) * (2 * BUFE);
    unsigned short* Bs = As + BUFE;
    int k0 = ch << 6;
#pragma unroll
    for (int i = 0; i < 2; ++i) {
      int flat = tid + (i << 8);               // 0..511
      int r = flat >> 3, c8 = (flat & 7) << 3; // element col (0,8,..,56)
      async_b128(As + r * LDAP + c8, Ablk + (size_t)r * lda + k0 + c8);
      if (BT)
        async_b128(Bs + r * LDAP + c8, Bblk + (size_t)r * ldb + k0 + c8);
      else
        async_b128(Bs + r * LDAP + c8, Bblk + (size_t)(k0 + r) * ldb + c8);
    }
  };

  stage(0);
  for (int ch = 0; ch < nch; ++ch) {
    wait_async0();
    __syncthreads();
    const unsigned short* As = smem + (ch & 1) * (2 * BUFE);
    const unsigned short* Bs = As + BUFE;
    if (ch + 1 < nch) stage(ch + 1);
#pragma unroll
    for (int ks = 0; ks < 64; ks += 32) {
      v16bf a = load_a16x32(As + rw * 16 * LDAP + ks, LDAP, lane);
      v16bf b0 = BT ? load_bt32x16(Bs + (cw + 0) * LDAP + ks, LDAP, lane)
                    : load_b32x16_rm(Bs + ks * LDAP + cw + 0, LDAP, lane);
      acc0 = wmma_bf16(a, b0, acc0);
      v16bf b1 = BT ? load_bt32x16(Bs + (cw + 16) * LDAP + ks, LDAP, lane)
                    : load_b32x16_rm(Bs + ks * LDAP + cw + 16, LDAP, lane);
      acc1 = wmma_bf16(a, b1, acc1);
    }
    __syncthreads();
  }

  int n = lane & 15, mb = (lane >> 4) << 3;
  size_t base0 = (size_t)(rw * 16 + mb) * ldc + cw + n;
  if (Cbf) {
#pragma unroll
    for (int j = 0; j < 8; ++j) {
      Cbf[base0 + (size_t)j * ldc] = f2bf(acc0[j] * scale);
      Cbf[base0 + (size_t)j * ldc + 16] = f2bf(acc1[j] * scale);
    }
  } else {
#pragma unroll
    for (int j = 0; j < 8; ++j) {
      Cf[base0 + (size_t)j * ldc] = acc0[j] * scale;
      Cf[base0 + (size_t)j * ldc + 16] = acc1[j] * scale;
    }
  }
}

// C[M,N] = A[M,K] * Bt[N,K]^T , 64x64 blocks
__global__ __launch_bounds__(256) void k_gemm_nt_lds(
    const unsigned short* __restrict__ A, const unsigned short* __restrict__ Bt,
    unsigned short* __restrict__ Cbf, float* __restrict__ Cf,
    int M, int N, int K, float scale) {
  __shared__ unsigned short smem[4 * BUFE];   // 2 bufs x (A + B), padded rows
  (void)M;
  int nb = N >> 6;
  int bm = blockIdx.x / nb, bn = blockIdx.x - bm * nb;
  unsigned short* cb = Cbf ? Cbf + (size_t)bm * 64 * N + bn * 64 : (unsigned short*)0;
  float* cf = Cf ? Cf + (size_t)bm * 64 * N + bn * 64 : (float*)0;
  gemm64<true>(A + (size_t)bm * 64 * K, K, Bt + (size_t)bn * 64 * K, K, K, scale, cb, cf, N, smem);
}

// ---------------- stat scores: per (b,h) 64x64 = q_stat * k_stat^T * 0.125 ----------------
__global__ __launch_bounds__(256) void k_stat_scores(
    const unsigned short* __restrict__ qs, const unsigned short* __restrict__ ks,
    float* __restrict__ scores) {
  int bh = blockIdx.x;                // b*8 + h
  int b = bh >> 3, h = bh & 7;
  int wave = threadIdx.x >> 5, lane = threadIdx.x & 31;
  const unsigned short* Abase = qs + (size_t)(b * QQ) * NHD + h * DHD;
  const unsigned short* Bbase = ks + (size_t)(b * SS) * NHD + h * DHD;
  for (int t = wave; t < 16; t += 8) {
    int tm = t >> 2, tn = t & 3;
    v8f c = vzero8();
    for (int k0 = 0; k0 < DHD; k0 += 32) {
      v16bf a = load_a16x32(Abase + (size_t)(tm * 16) * NHD + k0, NHD, lane);
      v16bf bt = load_bt32x16(Bbase + (size_t)(tn * 16) * NHD + k0, NHD, lane);
      c = wmma_bf16(a, bt, c);
    }
    int n = lane & 15, mb = (lane >> 4) * 8;
    float* out = scores + ((size_t)bh * QQ + tm * 16 + mb) * SS + tn * 16 + n;
#pragma unroll
    for (int j = 0; j < 8; ++j) out[(size_t)j * SS] = c[j] * 0.125f;
  }
}

// ---------------- exact top-8 keep + softmax over segments ----------------
__global__ void k_topk_softmax(const float* __restrict__ scores, const int* __restrict__ vlen,
                               float* __restrict__ statw) {
  int r = blockIdx.x * blockDim.x + threadIdx.x;    // row over B*H*Q = 4096
  if (r >= BB * HH * QQ) return;
  int b = r >> 9;
  int vl = vlen[b];
  const float* s = scores + (size_t)r * SS;

  float top[8];
#pragma unroll
  for (int i = 0; i < 8; ++i) top[i] = -3.0e38f;
  for (int si = 0; si < SS; ++si) {
    float x = (si < vl) ? s[si] : NEGV;
    if (x > top[7]) {
      int j = 7;
      while (j > 0 && top[j - 1] < x) { top[j] = top[j - 1]; --j; }
      top[j] = x;
    }
  }
  float thr = top[7];
  int cgt = 0;
  for (int si = 0; si < SS; ++si) {
    float x = (si < vl) ? s[si] : NEGV;
    if (x > thr) ++cgt;
  }
  int budget0 = 8 - cgt;

  float mx = -3.0e38f;
  int bd = budget0;
  for (int si = 0; si < SS; ++si) {
    float x = (si < vl) ? s[si] : NEGV;
    bool keep = (x > thr);
    if (!keep && x == thr && bd > 0) { keep = true; --bd; }
    if (keep && x > mx) mx = x;
  }
  float sum = 0.0f;
  bd = budget0;
  for (int si = 0; si < SS; ++si) {
    float x = (si < vl) ? s[si] : NEGV;
    bool keep = (x > thr);
    if (!keep && x == thr && bd > 0) { keep = true; --bd; }
    if (keep) sum += __expf(x - mx);
  }
  float inv = 1.0f / sum;
  bd = budget0;
  float* w = statw + (size_t)r * SS;
  for (int si = 0; si < SS; ++si) {
    float x = (si < vl) ? s[si] : NEGV;
    bool keep = (x > thr);
    if (!keep && x == thr && bd > 0) { keep = true; --bd; }
    w[si] = keep ? __expf(x - mx) * inv : 0.0f;
  }
}

// ---------------- token scores (last16) + softmax + combine with stat_w ----------------
__global__ __launch_bounds__(128) void k_tok_combine(
    const unsigned short* __restrict__ qt, const unsigned short* __restrict__ kt,
    const float* __restrict__ statw, unsigned short* __restrict__ comb) {
  __shared__ float sc[QQ * LAST];    // 64x16 fp32
  int id = blockIdx.x;               // (b*H + h)*S + s
  int s = id & 63;
  int bh = id >> 6;
  int b = bh >> 3, h = bh & 7;
  int wave = threadIdx.x >> 5, lane = threadIdx.x & 31;

  const unsigned short* Ab = qt + (size_t)(b * QQ + wave * 16) * NHD + h * DHD;
  const unsigned short* Bb = kt + (size_t)((b * SS + s) * LAST) * NHD + h * DHD;
  v8f c = vzero8();
  for (int k0 = 0; k0 < DHD; k0 += 32) {
    v16bf a = load_a16x32(Ab + k0, NHD, lane);
    v16bf bt = load_bt32x16(Bb + k0, NHD, lane);
    c = wmma_bf16(a, bt, c);
  }
  int n = lane & 15, mb = (lane >> 4) * 8;
#pragma unroll
  for (int j = 0; j < 8; ++j) sc[(wave * 16 + mb + j) * LAST + n] = c[j] * 0.125f;
  __syncthreads();

  int q = threadIdx.x;
  if (q < QQ) {
    float mx = -3.0e38f;
#pragma unroll
    for (int t = 0; t < LAST; ++t) mx = fmaxf(mx, sc[q * LAST + t]);
    float e[LAST];
    float sum = 0.0f;
#pragma unroll
    for (int t = 0; t < LAST; ++t) { e[t] = __expf(sc[q * LAST + t] - mx); sum += e[t]; }
    float w = statw[((size_t)bh * QQ + q) * SS + s] / sum;
    unsigned short* out = comb + ((size_t)bh * QQ + q) * (SS * LAST) + s * LAST;
#pragma unroll
    for (int t = 0; t < LAST; ++t) out[t] = f2bf(e[t] * w);
  }
}

// ---------------- attend: per (b,h): attn[64,64] = comb[64,1024] * Vp[1024,64] ----------------
__global__ __launch_bounds__(256) void k_attend_lds(
    const unsigned short* __restrict__ comb, const unsigned short* __restrict__ vp,
    unsigned short* __restrict__ attn) {
  __shared__ unsigned short smem[4 * BUFE];
  int bh = blockIdx.x;
  int b = bh >> 3, h = bh & 7;
  const int KK = SS * LAST;          // 1024
  gemm64<false>(comb + (size_t)bh * QQ * KK, KK,
                vp + (size_t)b * KK * NHD + h * DHD, NHD,
                KK, 1.0f,
                attn + (size_t)b * QQ * NHD + h * DHD, (float*)0, NHD, smem);
}

// ---------------- launch ----------------

extern "C" void kernel_launch(void* const* d_in, const int* in_sizes, int n_in,
                              void* d_out, int out_size, void* d_ws, size_t ws_size,
                              hipStream_t stream) {
  (void)in_sizes; (void)n_in; (void)out_size; (void)ws_size;
  const float* queries    = (const float*)d_in[0];
  const float* stat_keys  = (const float*)d_in[1];
  const float* token_keys = (const float*)d_in[2];
  const float* values     = (const float*)d_in[3];
  const int*   vlen       = (const int*)d_in[4];
  const float* W[6] = {(const float*)d_in[5], (const float*)d_in[6], (const float*)d_in[7],
                       (const float*)d_in[8], (const float*)d_in[9], (const float*)d_in[10]};

  char* ws = (char*)d_ws;
  size_t off = 0;
  auto take = [&](size_t bytes) -> void* {
    void* p = ws + off;
    off += (bytes + 255) & ~(size_t)255;
    return p;
  };
  unsigned short* Wt[6];
  for (int i = 0; i < 6; ++i) Wt[i] = (unsigned short*)take(512 * 512 * 2);
  unsigned short* q_bf  = (unsigned short*)take(512 * 512 * 2);
  unsigned short* sk_bf = (unsigned short*)take(512 * 512 * 2);
  unsigned short* tk_bf = (unsigned short*)take((size_t)8192 * 512 * 2);
  unsigned short* tv_bf = (unsigned short*)take((size_t)8192 * 512 * 2);
  unsigned short* qs_bf = (unsigned short*)take(512 * 512 * 2);
  unsigned short* qt_bf = (unsigned short*)take(512 * 512 * 2);
  unsigned short* ks_bf = (unsigned short*)take(512 * 512 * 2);
  unsigned short* kt_bf = (unsigned short*)take((size_t)8192 * 512 * 2);
  unsigned short* vp_bf = (unsigned short*)take((size_t)8192 * 512 * 2);
  float* scores = (float*)take((size_t)4096 * 64 * 4);
  float* statw  = (float*)take((size_t)4096 * 64 * 4);
  unsigned short* comb = (unsigned short*)take((size_t)4096 * 1024 * 2);
  unsigned short* attn = (unsigned short*)take(512 * 512 * 2);

  // 1) precision conversion (+ weight transpose, + last-16 token slice extraction)
  for (int i = 0; i < 6; ++i) k_weight_t<<<1024, 256, 0, stream>>>(W[i], Wt[i]);
  k_cvt<<<1024, 256, 0, stream>>>(queries, q_bf, 512 * 512);
  k_cvt<<<1024, 256, 0, stream>>>(stat_keys, sk_bf, 512 * 512);
  k_last16<<<16384, 256, 0, stream>>>(token_keys, tk_bf);
  k_last16<<<16384, 256, 0, stream>>>(values, tv_bf);

  // 2) projections: 64x64 macro-tiles, async-to-LDS double buffered, bf16 WMMA
  k_gemm_nt_lds<<<64, 256, 0, stream>>>(q_bf, Wt[0], qs_bf, nullptr, 512, 512, 512, 1.0f);
  k_gemm_nt_lds<<<64, 256, 0, stream>>>(q_bf, Wt[1], qt_bf, nullptr, 512, 512, 512, 1.0f);
  k_gemm_nt_lds<<<64, 256, 0, stream>>>(sk_bf, Wt[2], ks_bf, nullptr, 512, 512, 512, 1.0f);
  k_gemm_nt_lds<<<1024, 256, 0, stream>>>(tk_bf, Wt[3], kt_bf, nullptr, 8192, 512, 512, 1.0f);
  k_gemm_nt_lds<<<1024, 256, 0, stream>>>(tv_bf, Wt[4], vp_bf, nullptr, 8192, 512, 512, 1.0f);

  // 3) stat scores -> top-8 keep + softmax
  k_stat_scores<<<64, 256, 0, stream>>>(qs_bf, ks_bf, scores);
  k_topk_softmax<<<16, 256, 0, stream>>>(scores, vlen, statw);

  // 4) token softmax (last 16) combined with stat weights -> dense comb matrix
  k_tok_combine<<<4096, 128, 0, stream>>>(qt_bf, kt_bf, statw, comb);

  // 5) attend over values (async-LDS GEMM), then output projection (fp32 out)
  k_attend_lds<<<64, 256, 0, stream>>>(comb, vp_bf, attn);
  k_gemm_nt_lds<<<64, 256, 0, stream>>>(attn, Wt[5], nullptr, (float*)d_out, 512, 512, 512, 1.0f);
}
